// GIN_28020366639701
// MI455X (gfx1250) — compile-verified, matrix operand order checked by reading, not run on hardware
//
#include <hip/hip_runtime.h>

#define N_NODES 100000
#define N_EDGES 1600000
#define DCH     128
#define NLAY    3
#define SCAN_B  256
#define N_TILES (N_NODES/16)          // 6250 exact

typedef __attribute__((ext_vector_type(16))) __bf16         v16bf;
typedef __attribute__((ext_vector_type(8)))  float          v8f;
typedef __attribute__((ext_vector_type(4)))  float          v4f;
typedef __attribute__((ext_vector_type(4)))  unsigned       v4u;
typedef __attribute__((ext_vector_type(4)))  unsigned short v4us;

union BF16x16 { v16bf v; unsigned short u[16]; v4u q[2]; };

// split fp32 into bf16 hi + bf16 lo (hi + lo ~ fp32-accurate when used as 3-term WMMA)
__device__ __forceinline__ void split_bf16(float x, unsigned short& h, unsigned short& l){
  __bf16 hb = (__bf16)x;
  h = __builtin_bit_cast(unsigned short, hb);
  __bf16 lb = (__bf16)(x - (float)hb);
  l = __builtin_bit_cast(unsigned short, lb);
}

// A/B-fragment channel mapping for v_wmma_*_16x16x32:
//   k = kt*32 + kk,  kk -> half hf = (kk>>3)&1,  element e = (kk&7) + 8*(kk>>4)
// Fragment-ordered storage index: ((tile*4 + kt)*32 + (hf*16 + m))*16 + e

// ---------------- CSR construction helpers ----------------
__global__ void k_zero_i32(int* p, int n){
  int i = blockIdx.x*blockDim.x + threadIdx.x;
  if (i < n) p[i] = 0;
}
__global__ void k_copy_i32(int* d, const int* s, int n){
  int i = blockIdx.x*blockDim.x + threadIdx.x;
  if (i < n) d[i] = s[i];
}
__global__ void k_hist(const int* __restrict__ dst, int* __restrict__ counts){
  int e = blockIdx.x*blockDim.x + threadIdx.x;
  if (e < N_EDGES) atomicAdd(&counts[dst[e]], 1);
}
__global__ void k_scan_blocks(const int* __restrict__ counts, int* __restrict__ offsets,
                              int* __restrict__ bsum, int n){
  __shared__ int tmp[SCAN_B];
  int t = threadIdx.x;
  int i = blockIdx.x*SCAN_B + t;
  int v = (i < n) ? counts[i] : 0;
  tmp[t] = v; __syncthreads();
  for (int d = 1; d < SCAN_B; d <<= 1){
    int u = (t >= d) ? tmp[t-d] : 0; __syncthreads();
    tmp[t] += u; __syncthreads();
  }
  if (i < n) offsets[i] = tmp[t] - v;               // exclusive within block
  if (t == SCAN_B-1) bsum[blockIdx.x] = tmp[t];
}
__global__ void k_scan_sums(int* bsum, int nb){
  __shared__ int tmp[512];
  int t = threadIdx.x;
  int v = (t < nb) ? bsum[t] : 0;
  tmp[t] = v; __syncthreads();
  for (int d = 1; d < 512; d <<= 1){
    int u = (t >= d) ? tmp[t-d] : 0; __syncthreads();
    tmp[t] += u; __syncthreads();
  }
  if (t < nb) bsum[t] = tmp[t] - v;                 // exclusive
}
__global__ void k_add_bsum(int* __restrict__ offsets, const int* __restrict__ bsum, int n){
  int i = blockIdx.x*blockDim.x + threadIdx.x;
  if (i < n) offsets[i] += bsum[i / SCAN_B];
  if (i == n) offsets[n] = N_EDGES;
}
__global__ void k_place(const int* __restrict__ src, const int* __restrict__ dst,
                        int* __restrict__ cursor, int* __restrict__ esorted){
  int e = blockIdx.x*blockDim.x + threadIdx.x;
  if (e < N_EDGES){
    int p = atomicAdd(&cursor[dst[e]], 1);
    esorted[p] = src[e];
  }
}

// ------- pack weights into WMMA B-fragment layout (bf16 hi/lo planes) -------
__global__ void k_pack_w(const float* __restrict__ W1, const float* __restrict__ W2,
                         unsigned short* __restrict__ wpk){
  int idx = blockIdx.x*blockDim.x + threadIdx.x;
  if (idx >= NLAY*2*16384) return;
  int layer = idx / (2*16384);
  int rem   = idx - layer*(2*16384);
  int mat   = rem >> 14;
  int t     = rem & 16383;            // fragment index: ((kt*8+nt)*32+lane)*16+e
  int e    = t & 15;
  int lane = (t >> 4) & 31;
  int tile = t >> 9;
  int kt = tile >> 3, nt = tile & 7;
  int hf = lane >> 4, m = lane & 15;
  int k = kt*32 + hf*8 + (e & 7) + ((e >> 3) << 4);   // inverse of e-mapping
  int n = nt*16 + m;
  const float* W = (mat == 0 ? W1 : W2) + (size_t)layer*16384;
  unsigned short hi, lo;
  split_bf16(W[k*DCH + n], hi, lo);
  size_t base = (size_t)(layer*2 + mat) * 2 * 16384;
  wpk[base + t]         = hi;
  wpk[base + 16384 + t] = lo;
}

// ------- aggregation: agg = x + sum_{j in N(i)} x_j, emitted as packed bf16 hi/lo
//         A-fragment planes so the WMMA kernel does zero conversion work -------
__global__ void k_aggregate_pack(const float* __restrict__ x, const int* __restrict__ offsets,
                                 const int* __restrict__ esorted,
                                 unsigned short* __restrict__ aggh,
                                 unsigned short* __restrict__ aggl){
  int gid  = blockIdx.x*blockDim.x + threadIdx.x;
  int node = gid >> 5;
  int lane = threadIdx.x & 31;
  if (node >= N_NODES) return;
  int beg = offsets[node], end = offsets[node+1];
  const int c = lane * 4;                                  // 4 channels per lane
  v4f acc = *(const v4f*)(x + (size_t)node*DCH + c);       // self term (eps = 0)
  for (int j0 = beg; j0 < end; j0 += 32){
    int idx = j0 + lane;
    int s = (idx < end) ? esorted[idx] : 0;
    int cnt = end - j0; if (cnt > 32) cnt = 32;
    for (int t = 0; t < cnt; ++t){
      int sj = __builtin_amdgcn_readlane(s, t);
      acc += *(const v4f*)(x + (size_t)sj*DCH + c);        // 512B coalesced per wave
    }
  }
  // fragment-ordered packed store
  int tile = node >> 4, m = node & 15;
  int kt = c >> 5;
  int kk = c & 31;
  int hf = (kk >> 3) & 1;
  int e0 = (kk & 7) + ((kk >> 4) << 3);                    // e0..e0+3 contiguous
  size_t idx = ((size_t)(tile*4 + kt)*32 + (hf*16 + m))*16 + e0;
  v4us hv, lv;
  #pragma unroll
  for (int i = 0; i < 4; ++i){
    unsigned short h, l; split_bf16(acc[i], h, l);
    hv[i] = h; lv[i] = l;
  }
  *(v4us*)(aggh + idx) = hv;                               // 8B stores
  *(v4us*)(aggl + idx) = lv;
}

// ---------------- fused MLP, all-WMMA data path, M-blocked x2 ----------------
__device__ __forceinline__ void load_frag(BF16x16& f, const unsigned short* __restrict__ w,
                                          size_t elem16_idx){
  const v4u* p = (const v4u*)(w + (elem16_idx << 4));
  f.q[0] = p[0]; f.q[1] = p[1];
}
__device__ __forceinline__ v8f wmma_bf16(const BF16x16& a, const BF16x16& b, v8f c){
  return __builtin_amdgcn_wmma_f32_16x16x32_bf16(false, a.v, false, b.v, (short)0, c, false, false);
}

__global__ void __launch_bounds__(32)
k_gin_mlp(const unsigned short* __restrict__ aggh, const unsigned short* __restrict__ aggl,
          const unsigned short* __restrict__ w1pk, const float* __restrict__ b1,
          const unsigned short* __restrict__ w2pk, const float* __restrict__ b2,
          float* __restrict__ out, int relu_out)
{
  __shared__ unsigned short h1h[2*4*32*16];   // fragment-ordered h1 hi plane (2 tiles)
  __shared__ unsigned short h1l[2*4*32*16];   // and lo plane (8KB each)
  const int lane = threadIdx.x;               // single wave per block
  const int hf = lane >> 4;
  const int m  = lane & 15;
  const int tile0 = blockIdx.x*2;             // two 16-row tiles per wave, 0..6249 exact

  // Phase-1 A fragments for both tiles: pure b128 loads from packed agg planes
  BF16x16 ahi[2][4], alo[2][4];
  #pragma unroll
  for (int tt = 0; tt < 2; ++tt)
    #pragma unroll
    for (int kt = 0; kt < 4; ++kt){
      size_t fi = (size_t)((tile0 + tt)*4 + kt)*32 + lane;
      load_frag(ahi[tt][kt], aggh, fi);
      load_frag(alo[tt][kt], aggl, fi);
    }

  // GEMM1: h1 = relu(agg @ W1 + b1); each B fragment feeds both row tiles,
  // two accumulators interleave so WMMAs pipeline without an accum RAW chain.
  #pragma unroll
  for (int nt = 0; nt < 8; ++nt){
    v8f acc[2] = {{}, {}};
    #pragma unroll
    for (int kt = 0; kt < 4; ++kt){
      BF16x16 bh, bl;
      load_frag(bh, w1pk,         (size_t)(kt*8 + nt)*32 + lane);
      load_frag(bl, w1pk + 16384, (size_t)(kt*8 + nt)*32 + lane);
      acc[0] = wmma_bf16(ahi[0][kt], bh, acc[0]);
      acc[1] = wmma_bf16(ahi[1][kt], bh, acc[1]);
      acc[0] = wmma_bf16(ahi[0][kt], bl, acc[0]);
      acc[1] = wmma_bf16(ahi[1][kt], bl, acc[1]);
      acc[0] = wmma_bf16(alo[0][kt], bh, acc[0]);
      acc[1] = wmma_bf16(alo[1][kt], bh, acc[1]);
    }
    float bias = b1[nt*16 + m];
    // C element (e, lane) is h1[row = e + 8*hf][col n = nt*16 + m];
    // store at the phase-2 A-fragment position for channel k = n.
    const int ktp = nt >> 1;
    const int hfp = m >> 3;
    const int ep  = (m & 7) + 8*(nt & 1);
    #pragma unroll
    for (int tt = 0; tt < 2; ++tt)
      #pragma unroll
      for (int e = 0; e < 8; ++e){
        float v = fmaxf(acc[tt][e] + bias, 0.0f);
        unsigned short h, l; split_bf16(v, h, l);
        int lanep = hfp*16 + (e + 8*hf);
        int idx = ((tt*4 + ktp)*32 + lanep)*16 + ep;
        h1h[idx] = h;
        h1l[idx] = l;
      }
  }
  __syncthreads();   // single-wave workgroup: lowers to a NOP-cost barrier

  // Phase-2 A fragments: b128 loads from LDS planes
  #pragma unroll
  for (int tt = 0; tt < 2; ++tt)
    #pragma unroll
    for (int kt = 0; kt < 4; ++kt){
      int fi = (tt*4 + kt)*32 + lane;
      const v4u* ph = (const v4u*)(h1h + (fi << 4));
      ahi[tt][kt].q[0] = ph[0]; ahi[tt][kt].q[1] = ph[1];
      const v4u* pl = (const v4u*)(h1l + (fi << 4));
      alo[tt][kt].q[0] = pl[0]; alo[tt][kt].q[1] = pl[1];
    }

  // GEMM2: out = h1 @ W2 + b2 (+ inter-layer ReLU)
  #pragma unroll
  for (int nt = 0; nt < 8; ++nt){
    v8f acc[2] = {{}, {}};
    #pragma unroll
    for (int kt = 0; kt < 4; ++kt){
      BF16x16 bh, bl;
      load_frag(bh, w2pk,         (size_t)(kt*8 + nt)*32 + lane);
      load_frag(bl, w2pk + 16384, (size_t)(kt*8 + nt)*32 + lane);
      acc[0] = wmma_bf16(ahi[0][kt], bh, acc[0]);
      acc[1] = wmma_bf16(ahi[1][kt], bh, acc[1]);
      acc[0] = wmma_bf16(ahi[0][kt], bl, acc[0]);
      acc[1] = wmma_bf16(ahi[1][kt], bl, acc[1]);
      acc[0] = wmma_bf16(alo[0][kt], bh, acc[0]);
      acc[1] = wmma_bf16(alo[1][kt], bh, acc[1]);
    }
    float bias = b2[nt*16 + m];
    #pragma unroll
    for (int tt = 0; tt < 2; ++tt)
      #pragma unroll
      for (int e = 0; e < 8; ++e){
        float v = acc[tt][e] + bias;
        if (relu_out) v = fmaxf(v, 0.0f);
        out[(size_t)((tile0 + tt)*16 + e + 8*hf)*DCH + nt*16 + m] = v;
      }
  }
}

extern "C" void kernel_launch(void* const* d_in, const int* in_sizes, int n_in,
                              void* d_out, int out_size, void* d_ws, size_t ws_size,
                              hipStream_t stream)
{
  (void)in_sizes; (void)n_in; (void)out_size; (void)ws_size;
  const float* x  = (const float*)d_in[0];
  const int*   ei = (const int*)  d_in[1];
  const float* W1 = (const float*)d_in[2];
  const float* b1 = (const float*)d_in[3];
  const float* W2 = (const float*)d_in[4];
  const float* b2 = (const float*)d_in[5];
  float* out = (float*)d_out;
  const int* src = ei;              // edge_index[0]
  const int* dst = ei + N_EDGES;    // edge_index[1]

  char* ws = (char*)d_ws;
  size_t off = 0;
  auto take = [&](size_t bytes){ void* p = ws + off; off = (off + bytes + 255) & ~(size_t)255; return p; };
  unsigned short* aggh    = (unsigned short*)take((size_t)N_NODES*DCH*sizeof(unsigned short));
  unsigned short* aggl    = (unsigned short*)take((size_t)N_NODES*DCH*sizeof(unsigned short));
  int*            esorted = (int*)           take((size_t)N_EDGES*sizeof(int));
  int*            offsets = (int*)           take((size_t)(N_NODES+1)*sizeof(int));
  int*            counts  = (int*)           take((size_t)N_NODES*sizeof(int));
  int*            cursor  = (int*)           take((size_t)N_NODES*sizeof(int));
  int*            bsum    = (int*)           take(512*sizeof(int));
  unsigned short* wpk     = (unsigned short*)take((size_t)NLAY*2*2*16384*sizeof(unsigned short));

  const int nb = (N_NODES + SCAN_B - 1) / SCAN_B;   // 391
  k_zero_i32   <<<nb, SCAN_B, 0, stream>>>(counts, N_NODES);
  k_hist       <<<(N_EDGES+255)/256, 256, 0, stream>>>(dst, counts);
  k_scan_blocks<<<nb, SCAN_B, 0, stream>>>(counts, offsets, bsum, N_NODES);
  k_scan_sums  <<<1, 512, 0, stream>>>(bsum, nb);
  k_add_bsum   <<<(N_NODES+1+255)/256, 256, 0, stream>>>(offsets, bsum, N_NODES);
  k_copy_i32   <<<nb, SCAN_B, 0, stream>>>(cursor, offsets, N_NODES);
  k_place      <<<(N_EDGES+255)/256, 256, 0, stream>>>(src, dst, cursor, esorted);
  k_pack_w     <<<(NLAY*2*16384+255)/256, 256, 0, stream>>>(W1, W2, wpk);

  const float* xc = x;
  for (int L = 0; L < NLAY; ++L){
    k_aggregate_pack<<<(N_NODES*32+255)/256, 256, 0, stream>>>(xc, offsets, esorted, aggh, aggl);
    const unsigned short* w1pk = wpk + (size_t)(L*2 + 0)*2*16384;
    const unsigned short* w2pk = wpk + (size_t)(L*2 + 1)*2*16384;
    k_gin_mlp<<<N_TILES/2, 32, 0, stream>>>(aggh, aggl, w1pk, b1 + L*DCH, w2pk, b2 + L*DCH,
                                            out, (L < NLAY-1) ? 1 : 0);
    xc = out;   // ping-pong through d_out; only final contents are checked
  }
}
